// PixelContrastiveLoss_88038239634229
// MI455X (gfx1250) — matrix operand order repaired and checked by reading, not compiled, not run on hardware
//
#include <hip/hip_runtime.h>
#include <hip/hip_bf16.h>
#include <math.h>

// ---------------- problem constants (match reference) ----------------
#define BQ 8
#define CQ 64
#define HQ 256
#define WQ 256
#define PQ (HQ * WQ)
#define KS 2048          // MAX_SAMPLES
#define RAD 10
#define FG_CNT 1024      // MAX_FG
#define RING_CNT 768     // HARD_BUDGET
#define BG_CNT 256       // RAND_BUDGET
#define INV_T (1.0f / 0.07f)
#define NTILE (KS / 16)  // 128 tiles per dim
#define WPB 4            // waves per block in GEMM kernel
#define NSUPER (NTILE / WPB)
#define TILE_BYTES (16 * CQ * 2)   // one 16-row fp16 tile = 2KB

typedef _Float16 v16h __attribute__((ext_vector_type(16)));
typedef _Float16 v8h  __attribute__((ext_vector_type(8)));
typedef float    v8f  __attribute__((ext_vector_type(8)));

// gfx1250 async global->LDS staging (ASYNCcnt-tracked, ISA 15.18.3 / CDNA5 ISA)
#define ASYNC_STAGE_B128(ldsoff, gaddr)                                   \
    asm volatile("global_load_async_to_lds_b128 %0, %1, off"              \
                 :: "v"(ldsoff), "v"(gaddr) : "memory")
#define WAIT_ASYNC(N) asm volatile("s_wait_asynccnt " #N ::: "memory")

__device__ __forceinline__ unsigned pcg(unsigned x) {
    x = x * 747796405u + 2891336453u;
    x = ((x >> ((x >> 28) + 4u)) ^ x) * 277803737u;
    return (x >> 22) ^ x;
}

// ---------------- 1a: horizontal fg dilation (radius 10) ----------------
__global__ void k_hdil(const int* __restrict__ labels, unsigned char* __restrict__ hdil) {
    int idx = blockIdx.x * blockDim.x + threadIdx.x;
    if (idx >= BQ * PQ) return;
    int b = idx / PQ, p = idx - b * PQ;
    int i = p >> 8, j = p & 255;
    const int* lb = labels + b * PQ + i * WQ;
    int jlo = j - RAD < 0 ? 0 : j - RAD;
    int jhi = j + RAD >= WQ ? WQ - 1 : j + RAD;
    int d = 0;
    for (int jj = jlo; jj <= jhi; ++jj) d |= (lb[jj] == 1);
    hdil[idx] = (unsigned char)d;
}

// ---------------- 1b: vertical dilation + categorize ----------------
// cat: 0 = fg, 1 = ring (dilated & !fg), 2 = bg-only
__global__ void k_vcat(const int* __restrict__ labels, const unsigned char* __restrict__ hdil,
                       unsigned char* __restrict__ cat) {
    int idx = blockIdx.x * blockDim.x + threadIdx.x;
    if (idx >= BQ * PQ) return;
    int b = idx / PQ, p = idx - b * PQ;
    int i = p >> 8, j = p & 255;
    int ilo = i - RAD < 0 ? 0 : i - RAD;
    int ihi = i + RAD >= HQ ? HQ - 1 : i + RAD;
    int d = 0;
    for (int ii = ilo; ii <= ihi; ++ii) d |= hdil[b * PQ + ii * WQ + j];
    int fg = (labels[idx] == 1);
    cat[idx] = fg ? 0 : (d ? 1 : 2);
}

// ---------------- 2a: per-(image,row) category counts ----------------
__global__ void k_rowcnt(const unsigned char* __restrict__ cat, int* __restrict__ rowcnt) {
    int idx = blockIdx.x * blockDim.x + threadIdx.x;
    if (idx >= BQ * HQ) return;
    const unsigned char* cp = cat + (size_t)idx * WQ;
    int c0 = 0, c1 = 0, c2 = 0;
    for (int j = 0; j < WQ; ++j) {
        int c = cp[j];
        c0 += (c == 0); c1 += (c == 1); c2 += (c == 2);
    }
    rowcnt[idx * 3 + 0] = c0;
    rowcnt[idx * 3 + 1] = c1;
    rowcnt[idx * 3 + 2] = c2;
}

// ---------------- 2b: exclusive scan over rows, per (image,cat) ----------------
__global__ void k_scan(const int* __restrict__ rowcnt, int* __restrict__ rowoff,
                       int* __restrict__ totals) {
    int t = threadIdx.x;
    if (t >= BQ * 3) return;
    int b = t / 3, c = t - b * 3;
    int run = 0;
    for (int r = 0; r < HQ; ++r) {
        int a = (b * HQ + r) * 3 + c;
        rowoff[a] = run;
        run += rowcnt[a];
    }
    totals[t] = run;
}

// ---------------- 3: deterministic strided+jittered sample selection ----------------
__global__ void k_select(const unsigned char* __restrict__ cat, const int* __restrict__ labels,
                         const int* __restrict__ rowoff, const int* __restrict__ totals,
                         int* __restrict__ sidx, int* __restrict__ ysArr) {
    int idx = blockIdx.x * blockDim.x + threadIdx.x;
    if (idx >= BQ * KS) return;
    int b = idx / KS, k = idx - b * KS;
    int cid, budget, kk;
    if (k < FG_CNT)                 { cid = 0; budget = FG_CNT;   kk = k; }
    else if (k < FG_CNT + RING_CNT) { cid = 1; budget = RING_CNT; kk = k - FG_CNT; }
    else                            { cid = 2; budget = BG_CNT;   kk = k - FG_CNT - RING_CNT; }
    long long total = totals[b * 3 + cid];
    int p = 0;
    if (total > 0) {
        long long r = (long long)kk * total / budget;
        long long stride = total / budget;
        if (stride > 1) r += (long long)(pcg((unsigned)idx * 2654435761u + 12345u) % (unsigned)stride);
        if (r > total - 1) r = total - 1;
        int lo = 0, hi = HQ - 1;
        while (lo < hi) {
            int mid = (lo + hi + 1) >> 1;
            if ((long long)rowoff[(b * HQ + mid) * 3 + cid] <= r) lo = mid; else hi = mid - 1;
        }
        long long cnt = r - rowoff[(b * HQ + lo) * 3 + cid];
        const unsigned char* cp = cat + (size_t)(b * HQ + lo) * WQ;
        int jsel = WQ - 1;
        for (int j = 0; j < WQ; ++j) {
            if (cp[j] == cid) { if (cnt == 0) { jsel = j; break; } --cnt; }
        }
        p = lo * WQ + jsel;
    }
    sidx[idx] = p;
    ysArr[idx] = labels[b * PQ + p];
}

// ---------------- 4: gather + L2 normalize -> fp16 [B][KS][CQ] ----------------
__global__ void k_gathernorm(const float* __restrict__ feats, const int* __restrict__ sidx,
                             _Float16* __restrict__ F) {
    int b = blockIdx.y, k = blockIdx.x, c = threadIdx.x;
    __shared__ float red[CQ];
    int p = sidx[b * KS + k];
    float v = feats[((size_t)b * CQ + c) * PQ + p];
    red[c] = v * v;
    __syncthreads();
    for (int s = CQ / 2; s > 0; s >>= 1) {
        if (c < s) red[c] += red[c + s];
        __syncthreads();
    }
    float nrm = sqrtf(red[0]);
    nrm = nrm > 1e-12f ? nrm : 1e-12f;
    F[((size_t)b * KS + k) * CQ + c] = (_Float16)(v / nrm);
}

// ---------------- 5: WMMA kernel: exp((F F^T - 1)/T) row sums ----------------
// Block = 4 waves; wave w owns output row-tile (blockIdx.x*WPB + w) and loops over
// all 128 j-tiles. B tiles are double-buffered in LDS via async global->LDS loads
// (ASYNCcnt), so the next 2KB tile streams in while the current WMMAs run.
// Row sums finish entirely in registers -> plain stores, no atomics, no pre-zero.
__global__ void __launch_bounds__(WPB * 32)
k_contrast(const _Float16* __restrict__ F, const int* __restrict__ ysArr,
           float* __restrict__ denom, float* __restrict__ possum) {
    __shared__ _Float16 sB[2][16][CQ];   // double-buffered 16-row B tile (2 x 2KB)

    int b = blockIdx.y;
    int t = threadIdx.x;                 // 0..127
    int wave = t >> 5;
    int lane = t & 31;
    int r16 = lane & 15, hi = lane >> 4;

    const _Float16* Fb = F + (size_t)b * KS * CQ;
    const int* ys_b = ysArr + b * KS;
    int i0 = (blockIdx.x * WPB + wave) * 16;

    // ---- A fragments for this wave's 16 output rows (persist in registers) ----
    // 16-bit A 16x32 layout: lanes 0-15 hold M=r16, K=0..7 & 16..23; lanes 16-31 K=8..15 & 24..31
    const _Float16* arow = Fb + (size_t)(i0 + r16) * CQ;
    v8h a0l = *(const v8h*)(arow + hi * 8);
    v8h a0h = *(const v8h*)(arow + 16 + hi * 8);
    v8h a1l = *(const v8h*)(arow + 32 + hi * 8);
    v8h a1h = *(const v8h*)(arow + 48 + hi * 8);
    v16h A0 = __builtin_shufflevector(a0l, a0h, 0,1,2,3,4,5,6,7,8,9,10,11,12,13,14,15);
    v16h A1 = __builtin_shufflevector(a1l, a1h, 0,1,2,3,4,5,6,7,8,9,10,11,12,13,14,15);

    // labels of this wave's rows (C/D layout: VGPR v -> row v + 8*hi)
    int ysi[8];
#pragma unroll
    for (int v = 0; v < 8; ++v) ysi[v] = ys_b[i0 + hi * 8 + v];

    float dAcc[8], pAcc[8];
#pragma unroll
    for (int v = 0; v < 8; ++v) { dAcc[v] = 0.0f; pAcc[v] = 0.0f; }

    unsigned ldsBase = (unsigned)(unsigned long long)(uintptr_t)&sB[0][0][0];

    // stage round 0 (each of the 4 waves async-loads 512B of the 2KB tile)
    {
        const char* g = (const char*)Fb + (size_t)t * 16;
        unsigned lds = ldsBase + (unsigned)t * 16u;
        ASYNC_STAGE_B128(lds, g);
    }

#pragma unroll 1
    for (int r = 0; r < NTILE; ++r) {
        if (r + 1 < NTILE) {
            // prefetch next tile into the other buffer
            const char* g = (const char*)Fb + (size_t)(r + 1) * TILE_BYTES + (size_t)t * 16;
            unsigned lds = ldsBase + (unsigned)(((r + 1) & 1) * TILE_BYTES) + (unsigned)t * 16u;
            ASYNC_STAGE_B128(lds, g);
            WAIT_ASYNC(0x1);   // in-order: round r's stage is complete
        } else {
            WAIT_ASYNC(0x0);
        }
        __syncthreads();       // tile r visible to all 4 waves

        // ---- B fragments from LDS (32x16 layout: lanes 0-15 K=0..7 & 16..23, lanes 16-31 K=8..15 & 24..31) ----
        const _Float16* brow = &sB[r & 1][r16][0];
        v8h b0l = *(const v8h*)(brow + hi * 8);
        v8h b0h = *(const v8h*)(brow + 16 + hi * 8);
        v8h b1l = *(const v8h*)(brow + 32 + hi * 8);
        v8h b1h = *(const v8h*)(brow + 48 + hi * 8);
        v16h B0 = __builtin_shufflevector(b0l, b0h, 0,1,2,3,4,5,6,7,8,9,10,11,12,13,14,15);
        v16h B1 = __builtin_shufflevector(b1l, b1h, 0,1,2,3,4,5,6,7,8,9,10,11,12,13,14,15);

        v8f acc = {};
        acc = __builtin_amdgcn_wmma_f32_16x16x32_f16(false, A0, false, B0, (short)0, acc, false, false);
        acc = __builtin_amdgcn_wmma_f32_16x16x32_f16(false, A1, false, B1, (short)0, acc, false, false);

        int gj = r * 16 + r16;
        int ysj = ys_b[gj];
#pragma unroll
        for (int v = 0; v < 8; ++v) {
            int gi = i0 + hi * 8 + v;
            // row max == 1/T (diagonal of cosine sim); exp(logit - max) = exp((dot-1)/T)
            float e = (gi == gj) ? 0.0f : __expf((acc[v] - 1.0f) * INV_T);
            dAcc[v] += e;
            pAcc[v] += (ysi[v] == ysj) ? e : 0.0f;
        }
        __syncthreads();       // everyone done reading buffer (r&1) before it is re-staged
    }

    // ---- final half-wave reduction (C/D cols live on 16 lanes) + direct stores ----
#pragma unroll
    for (int v = 0; v < 8; ++v) {
        float e = dAcc[v], pe = pAcc[v];
#pragma unroll
        for (int m = 1; m < 16; m <<= 1) {
            e  += __shfl_xor(e,  m, 16);
            pe += __shfl_xor(pe, m, 16);
        }
        if (r16 == 0) {
            int gi = i0 + hi * 8 + v;
            denom[b * KS + gi]  = e;
            possum[b * KS + gi] = pe;
        }
    }
}

// ---------------- 6: per-image loss reduction ----------------
__global__ void k_imgloss(const float* __restrict__ denom, const float* __restrict__ possum,
                          const int* __restrict__ ysArr, float* __restrict__ imgloss) {
    int b = blockIdx.x, t = threadIdx.x;
    __shared__ float ssum[256];
    __shared__ int scnt[256];
    float s = 0.0f; int c = 0;
    for (int i = t; i < KS; i += 256) {
        if (ysArr[b * KS + i] == 1) {
            float ps = possum[b * KS + i];
            if (ps > 0.0f) {                       // pos.any(axis=1)
                float dn = denom[b * KS + i];
                ps = ps > 1e-8f ? ps : 1e-8f;
                dn = dn > 1e-8f ? dn : 1e-8f;
                s += -__logf(ps / dn);
                c += 1;
            }
        }
    }
    ssum[t] = s; scnt[t] = c;
    __syncthreads();
    for (int st = 128; st > 0; st >>= 1) {
        if (t < st) { ssum[t] += ssum[t + st]; scnt[t] += scnt[t + st]; }
        __syncthreads();
    }
    if (t == 0) {
        int n = scnt[0] > 0 ? scnt[0] : 1;
        imgloss[b] = ssum[0] / (float)n;
    }
}

// ---------------- 7: batch mean ----------------
__global__ void k_final(const float* __restrict__ imgloss, float* __restrict__ out) {
    if (blockIdx.x == 0 && threadIdx.x == 0) {
        float s = 0.0f;
        for (int b = 0; b < BQ; ++b) s += imgloss[b];
        out[0] = s / (float)BQ;
    }
}

extern "C" void kernel_launch(void* const* d_in, const int* in_sizes, int n_in,
                              void* d_out, int out_size, void* d_ws, size_t ws_size,
                              hipStream_t stream) {
    const float* feats = (const float*)d_in[0];   // [B, C, H, W] fp32
    const int* labels  = (const int*)d_in[1];     // [B, H, W] int32
    float* out = (float*)d_out;                   // scalar fp32

    char* ws = (char*)d_ws;
    size_t off = 0;
    auto alloc = [&](size_t bytes) -> char* {
        char* p = ws + off;
        off = (off + bytes + 255) & ~(size_t)255;
        return p;
    };
    unsigned char* cat  = (unsigned char*)alloc((size_t)BQ * PQ);
    unsigned char* hdil = (unsigned char*)alloc((size_t)BQ * PQ);
    int* rowcnt = (int*)alloc((size_t)BQ * HQ * 3 * sizeof(int));
    int* rowoff = (int*)alloc((size_t)BQ * HQ * 3 * sizeof(int));
    int* totals = (int*)alloc((size_t)BQ * 3 * sizeof(int));
    int* sidx   = (int*)alloc((size_t)BQ * KS * sizeof(int));
    int* ysArr  = (int*)alloc((size_t)BQ * KS * sizeof(int));
    _Float16* F = (_Float16*)alloc((size_t)BQ * KS * CQ * sizeof(_Float16));
    float* denom  = (float*)alloc((size_t)BQ * KS * sizeof(float));
    float* possum = (float*)alloc((size_t)BQ * KS * sizeof(float));
    float* imgloss = (float*)alloc((size_t)BQ * sizeof(float));
    (void)ws_size; (void)in_sizes; (void)n_in; (void)out_size;

    int npix = BQ * PQ;
    k_hdil<<<(npix + 255) / 256, 256, 0, stream>>>(labels, hdil);
    k_vcat<<<(npix + 255) / 256, 256, 0, stream>>>(labels, hdil, cat);
    k_rowcnt<<<(BQ * HQ + 255) / 256, 256, 0, stream>>>(cat, rowcnt);
    k_scan<<<1, 32, 0, stream>>>(rowcnt, rowoff, totals);
    k_select<<<(BQ * KS + 255) / 256, 256, 0, stream>>>(cat, labels, rowoff, totals, sidx, ysArr);
    k_gathernorm<<<dim3(KS, BQ), CQ, 0, stream>>>(feats, sidx, F);
    k_contrast<<<dim3(NSUPER, BQ), WPB * 32, 0, stream>>>(F, ysArr, denom, possum);
    k_imgloss<<<BQ, 256, 0, stream>>>(denom, possum, ysArr, imgloss);
    k_final<<<1, 32, 0, stream>>>(imgloss, out);
}
